// DecoderBlock_41128606826523
// MI455X (gfx1250) — compile-verified, matrix operand order checked by reading, not run on hardware
//
#include <hip/hip_runtime.h>
#include <hip/hip_bf16.h>

// ---------------------------------------------------------------------------
// Types for CDNA5 WMMA (wave32): bf16 16x16x32, f32 accumulate.
// ---------------------------------------------------------------------------
typedef __attribute__((ext_vector_type(16))) __bf16          v16bf;
typedef __attribute__((ext_vector_type(16))) unsigned short  u16x16;
typedef __attribute__((ext_vector_type(8)))  unsigned short  u16x8;
typedef __attribute__((ext_vector_type(8)))  float           v8f;

#define D_MODEL 1024
#define N_HEADS 16
#define HEAD_DIM 64
#define D_FF    4096
#define SEQ_T   2048
#define BATCH   2
#define M_ROWS  (BATCH * SEQ_T)   // 4096
#define QKV_STRIDE (3 * D_MODEL)  // 3072

__device__ __forceinline__ unsigned short f2bf(float f) {
    unsigned int u = __float_as_uint(f);
    u += 0x7FFFu + ((u >> 16) & 1u);   // round-to-nearest-even
    return (unsigned short)(u >> 16);
}

__device__ __forceinline__ v16bf frag_from_halves(u16x8 lo, u16x8 hi) {
    u16x16 t;
#pragma unroll
    for (int i = 0; i < 8; ++i) { t[i] = lo[i]; t[i + 8] = hi[i]; }
    return __builtin_bit_cast(v16bf, t);
}

// Load a 16x32 bf16 A/B fragment: per-lane element i <-> K = i+8*((i>=8)+(lane>=16))
__device__ __forceinline__ v16bf load_frag(const unsigned short* p) {
    return frag_from_halves(*(const u16x8*)p, *(const u16x8*)(p + 16));
}

// ---- CDNA5 async global->LDS (ASYNCcnt) + explicit counter fences ---------
__device__ __forceinline__ void async_load_b128(unsigned lds_off, const void* gaddr) {
    asm volatile("global_load_async_to_lds_b128 %0, %1, off"
                 :: "v"(lds_off), "v"(gaddr) : "memory");
}
__device__ __forceinline__ void wait_async0() {
    asm volatile("s_wait_asynccnt 0" ::: "memory");
}
__device__ __forceinline__ void wait_ds0() {
    asm volatile("s_wait_dscnt 0" ::: "memory");
}

// ---------------------------------------------------------------------------
// fp32 [K,N] weight -> bf16 [N,K] transposed (B-frags become 16B loads)
// ---------------------------------------------------------------------------
__global__ void transpose_to_bf16_kernel(const float* __restrict__ W,
                                         unsigned short* __restrict__ Wt,
                                         int K, int N) {
    size_t idx = (size_t)blockIdx.x * blockDim.x + threadIdx.x;
    if (idx >= (size_t)K * N) return;
    int k = (int)(idx / N);
    int n = (int)(idx % N);
    Wt[(size_t)n * K + k] = f2bf(W[idx]);
}

// ---------------------------------------------------------------------------
// LayerNorm (fp32 in) -> bf16 out. One block per row.
// ---------------------------------------------------------------------------
__global__ void layernorm_bf16_kernel(const float* __restrict__ x,
                                      const float* __restrict__ gamma,
                                      const float* __restrict__ beta,
                                      unsigned short* __restrict__ out,
                                      int D) {
    int row = blockIdx.x;
    const float* xr = x + (size_t)row * D;
    float s = 0.f, s2 = 0.f;
    for (int i = threadIdx.x; i < D; i += blockDim.x) {
        float v = xr[i]; s += v; s2 += v * v;
    }
#pragma unroll
    for (int off = 16; off > 0; off >>= 1) {
        s  += __shfl_xor(s,  off, 32);
        s2 += __shfl_xor(s2, off, 32);
    }
    __shared__ float rs[8], rs2[8];
    __shared__ float fmu, finv;
    int wave = threadIdx.x >> 5, lane = threadIdx.x & 31;
    if (lane == 0) { rs[wave] = s; rs2[wave] = s2; }
    __syncthreads();
    if (threadIdx.x == 0) {
        float ts = 0.f, ts2 = 0.f;
#pragma unroll
        for (int i = 0; i < 8; ++i) { ts += rs[i]; ts2 += rs2[i]; }
        float mu  = ts / (float)D;
        float var = ts2 / (float)D - mu * mu;
        fmu = mu; finv = rsqrtf(var + 1e-5f);
    }
    __syncthreads();
    float mu = fmu, inv = finv;
    for (int i = threadIdx.x; i < D; i += blockDim.x) {
        float v = (xr[i] - mu) * inv * gamma[i] + beta[i];
        out[(size_t)row * D + i] = f2bf(v);
    }
}

// ---------------------------------------------------------------------------
// Register-blocked WMMA GEMM: C[M,N] = A[M,K](bf16) x Bt[N,K](bf16) + bias
// Wave computes 32x64 (2 M-tiles x 4 N-tiles): 6 frag loads -> 8 WMMAs per
// 32-wide K step (~43 FLOP/byte from L2). Block = 4 waves (2x2) = 64x128 tile.
// EPI 0: -> bf16    EPI 1: GELU -> bf16    EPI 2: +residual -> f32
// ---------------------------------------------------------------------------
template <int EPI>
__global__ __launch_bounds__(128)
void gemm_wmma_kernel(const unsigned short* __restrict__ A,
                      const unsigned short* __restrict__ Bt,
                      const float* __restrict__ bias,
                      const float* __restrict__ resid,
                      void* __restrict__ out,
                      int M, int N, int K) {
    int lane  = threadIdx.x & 31;
    int wave  = threadIdx.x >> 5;
    int wm    = wave >> 1;                    // 0..1
    int wn    = wave & 1;                     // 0..1
    int tm0   = blockIdx.y * 4 + wm * 2;      // first of 2 M-tiles
    int tn0   = blockIdx.x * 8 + wn * 4;      // first of 4 N-tiles
    int lrow  = lane & 15;
    int lhalf = lane >> 4;
    int kh    = lhalf * 8;

    const unsigned short* ap0 = A + (size_t)(tm0 * 16 + lrow) * K + kh;
    const unsigned short* ap1 = ap0 + (size_t)16 * K;
    const unsigned short* bp0 = Bt + (size_t)((tn0 + 0) * 16 + lrow) * K + kh;
    const unsigned short* bp1 = Bt + (size_t)((tn0 + 1) * 16 + lrow) * K + kh;
    const unsigned short* bp2 = Bt + (size_t)((tn0 + 2) * 16 + lrow) * K + kh;
    const unsigned short* bp3 = Bt + (size_t)((tn0 + 3) * 16 + lrow) * K + kh;

    v8f acc[2][4] = {};
    for (int kb = 0; kb < K; kb += 32) {
        v16bf a0 = load_frag(ap0 + kb);
        v16bf a1 = load_frag(ap1 + kb);
        v16bf b0 = load_frag(bp0 + kb);
        v16bf b1 = load_frag(bp1 + kb);
        v16bf b2 = load_frag(bp2 + kb);
        v16bf b3 = load_frag(bp3 + kb);
        acc[0][0] = __builtin_amdgcn_wmma_f32_16x16x32_bf16(false, a0, false, b0, (short)0, acc[0][0], false, false);
        acc[0][1] = __builtin_amdgcn_wmma_f32_16x16x32_bf16(false, a0, false, b1, (short)0, acc[0][1], false, false);
        acc[0][2] = __builtin_amdgcn_wmma_f32_16x16x32_bf16(false, a0, false, b2, (short)0, acc[0][2], false, false);
        acc[0][3] = __builtin_amdgcn_wmma_f32_16x16x32_bf16(false, a0, false, b3, (short)0, acc[0][3], false, false);
        acc[1][0] = __builtin_amdgcn_wmma_f32_16x16x32_bf16(false, a1, false, b0, (short)0, acc[1][0], false, false);
        acc[1][1] = __builtin_amdgcn_wmma_f32_16x16x32_bf16(false, a1, false, b1, (short)0, acc[1][1], false, false);
        acc[1][2] = __builtin_amdgcn_wmma_f32_16x16x32_bf16(false, a1, false, b2, (short)0, acc[1][2], false, false);
        acc[1][3] = __builtin_amdgcn_wmma_f32_16x16x32_bf16(false, a1, false, b3, (short)0, acc[1][3], false, false);
    }

    // C/D layout: element i -> row = tile*16 + 8*lhalf + i, col = tile*16 + lrow
#pragma unroll
    for (int mi = 0; mi < 2; ++mi) {
        int rb = (tm0 + mi) * 16 + 8 * lhalf;
#pragma unroll
        for (int ni = 0; ni < 4; ++ni) {
            int c = (tn0 + ni) * 16 + lrow;
            float bv = bias[c];
#pragma unroll
            for (int i = 0; i < 8; ++i) {
                int r = rb + i;
                float v = acc[mi][ni][i] + bv;
                if (EPI == 1) {
                    v = 0.5f * v * (1.0f + erff(v * 0.70710678118654752f));
                }
                if (EPI == 2) {
                    ((float*)out)[(size_t)r * N + c] = v + resid[(size_t)r * N + c];
                } else {
                    ((unsigned short*)out)[(size_t)r * N + c] = f2bf(v);
                }
            }
        }
    }
}

// ---------------------------------------------------------------------------
// Causal flash attention. qkv bf16: [B*T, 3*D] rows laid out [q|k|v], head
// dims contiguous. One wave per (batch, head, 16-query tile). V tiles are
// staged into per-wave LDS with async global->LDS b128 copies (ASYNCcnt),
// pipelined one key-block ahead; P goes through LDS to re-shape C->A layout
// (DS ops are in-order per wave; s_wait_dscnt 0 fences write->read).
// ---------------------------------------------------------------------------
__global__ __launch_bounds__(128)
void attention_kernel(const unsigned short* __restrict__ qkv,
                      unsigned short* __restrict__ attn_out) {
    __shared__ alignas(16) unsigned short plds[4][16 * 32];       // P tiles
    __shared__ alignas(16) unsigned short vlds[4][32 * HEAD_DIM]; // V tiles

    int lane  = threadIdx.x & 31;
    int wave  = threadIdx.x >> 5;
    int task  = blockIdx.x * 4 + wave;           // 4096 tasks
    int qt    = task & 127;                      // query tile (T/16 = 128)
    int h     = (task >> 7) & 15;
    int b     = task >> 11;
    int lrow  = lane & 15;
    int lhalf = lane >> 4;
    int kh    = lhalf * 8;

    const unsigned short* base  = qkv + (size_t)b * SEQ_T * QKV_STRIDE;
    const unsigned short* vbase = base + 2 * D_MODEL + h * HEAD_DIM;
    // Raw 32-bit LDS offset (low 32 bits of the shared-aperture address).
    unsigned vlds_off = (unsigned)(uintptr_t)&vlds[wave][0];

    // Stage V key-block kb into this wave's LDS tile: 8 coalesced 16B/lane
    // async transactions cover the 32x64 bf16 tile.
    auto stage_V = [&](int kb) {
#pragma unroll
        for (int cc = 0; cc < 8; ++cc) {
            int id  = cc * 32 + lane;            // 16B chunk index
            int key = id >> 3;                   // 0..31
            int dim = (id & 7) * 8;              // 0..56
            async_load_b128(vlds_off + (unsigned)id * 16,
                            vbase + (size_t)(kb + key) * QKV_STRIDE + dim);
        }
    };

    // Q fragments for d = 0..31 and 32..63
    v16bf qf[2];
    {
        const unsigned short* qp =
            base + (size_t)(qt * 16 + lrow) * QKV_STRIDE + h * HEAD_DIM;
        qf[0] = load_frag(qp + kh);
        qf[1] = load_frag(qp + 32 + kh);
    }

    float m[8], l[8], sc[8];
    v8f o[4] = {};
#pragma unroll
    for (int i = 0; i < 8; ++i) { m[i] = -1e30f; l[i] = 0.f; }

    int qend = (qt + 1) * 16;                    // last needed key (excl.)
    stage_V(0);                                  // pipeline prologue

    for (int kb = 0; kb < qend; kb += 32) {
        // ---- S = Q K^T for 32 keys (4 WMMAs) --------------------------------
        v8f s0 = {}, s1 = {};
#pragma unroll
        for (int ds = 0; ds < 2; ++ds) {
            const unsigned short* kp = base + (size_t)(kb + lrow) * QKV_STRIDE +
                                       D_MODEL + h * HEAD_DIM + ds * 32 + kh;
            v16bf kf0 = load_frag(kp);
            s0 = __builtin_amdgcn_wmma_f32_16x16x32_bf16(
                false, qf[ds], false, kf0, (short)0, s0, false, false);
            v16bf kf1 = load_frag(kp + (size_t)16 * QKV_STRIDE);
            s1 = __builtin_amdgcn_wmma_f32_16x16x32_bf16(
                false, qf[ds], false, kf1, (short)0, s1, false, false);
        }

        // ---- online softmax (rows live in 16-lane halves) ------------------
        int rbase = qt * 16 + lhalf * 8;
#pragma unroll
        for (int i = 0; i < 8; ++i) {
            int r = rbase + i;
            float a0 = s0[i] * 0.125f;           // 1/sqrt(64)
            float a1 = s1[i] * 0.125f;
            if (kb + lrow > r)       a0 = -1e30f;
            if (kb + 16 + lrow > r)  a1 = -1e30f;
            float mx = fmaxf(a0, a1);
#pragma unroll
            for (int off = 1; off < 16; off <<= 1)
                mx = fmaxf(mx, __shfl_xor(mx, off, 32));
            float mnew = fmaxf(m[i], mx);
            float p0 = __expf(a0 - mnew);
            float p1 = __expf(a1 - mnew);
            float rs = p0 + p1;
#pragma unroll
            for (int off = 1; off < 16; off <<= 1)
                rs += __shfl_xor(rs, off, 32);
            float corr = __expf(m[i] - mnew);
            l[i] = l[i] * corr + rs;
            m[i] = mnew;
            sc[i] = corr;
            unsigned short* pl = &plds[wave][(lhalf * 8 + i) * 32];
            pl[lrow]      = f2bf(p0);
            pl[lrow + 16] = f2bf(p1);
        }

        // Drain DS writes (P) and async V staging before reading either.
        wait_ds0();
        wait_async0();

        v16bf pf = load_frag(&plds[wave][lrow * 32 + kh]);

#pragma unroll
        for (int nt = 0; nt < 4; ++nt) {
#pragma unroll
            for (int i = 0; i < 8; ++i) o[nt][i] *= sc[i];
        }
#pragma unroll
        for (int nt = 0; nt < 4; ++nt) {
            u16x16 vt;
#pragma unroll
            for (int i = 0; i < 16; ++i) {
                int key = i + 8 * ((i >> 3) + lhalf);
                vt[i] = vlds[wave][key * HEAD_DIM + nt * 16 + lrow];
            }
            o[nt] = __builtin_amdgcn_wmma_f32_16x16x32_bf16(
                false, pf, false, __builtin_bit_cast(v16bf, vt),
                (short)0, o[nt], false, false);
        }

        // Next V tile: fence pending DS reads, then re-issue async writes.
        if (kb + 32 < qend) {
            wait_ds0();
            stage_V(kb + 32);
        }
    }

#pragma unroll
    for (int nt = 0; nt < 4; ++nt) {
#pragma unroll
        for (int i = 0; i < 8; ++i) {
            int r = b * SEQ_T + qt * 16 + lhalf * 8 + i;
            attn_out[(size_t)r * D_MODEL + h * HEAD_DIM + nt * 16 + lrow] =
                f2bf(o[nt][i] / l[i]);
        }
    }
}

// ---------------------------------------------------------------------------
// Host-side orchestration
// ---------------------------------------------------------------------------
extern "C" void kernel_launch(void* const* d_in, const int* in_sizes, int n_in,
                              void* d_out, int out_size, void* d_ws, size_t ws_size,
                              hipStream_t stream) {
    (void)in_sizes; (void)n_in; (void)out_size; (void)ws_size;
    const float* x     = (const float*)d_in[0];
    const float* ln1_g = (const float*)d_in[1];
    const float* ln1_b = (const float*)d_in[2];
    const float* ln2_g = (const float*)d_in[3];
    const float* ln2_b = (const float*)d_in[4];
    const float* w_qkv = (const float*)d_in[5];
    const float* b_qkv = (const float*)d_in[6];
    const float* w_out = (const float*)d_in[7];
    const float* b_out = (const float*)d_in[8];
    const float* w_ff1 = (const float*)d_in[9];
    const float* b_ff1 = (const float*)d_in[10];
    const float* w_ff2 = (const float*)d_in[11];
    const float* b_ff2 = (const float*)d_in[12];

    char* ws = (char*)d_ws;
    unsigned short* h1    = (unsigned short*)ws; ws += (size_t)M_ROWS * D_MODEL * 2;
    unsigned short* wqkvT = (unsigned short*)ws; ws += (size_t)(3*D_MODEL) * D_MODEL * 2;
    unsigned short* qkvb  = (unsigned short*)ws; ws += (size_t)M_ROWS * (3*D_MODEL) * 2;
    unsigned short* attnb = (unsigned short*)ws; ws += (size_t)M_ROWS * D_MODEL * 2;
    unsigned short* woutT = (unsigned short*)ws; ws += (size_t)D_MODEL * D_MODEL * 2;
    float*          x1    = (float*)ws;          ws += (size_t)M_ROWS * D_MODEL * 4;
    unsigned short* h2    = (unsigned short*)ws; ws += (size_t)M_ROWS * D_MODEL * 2;
    unsigned short* wff1T = (unsigned short*)ws; ws += (size_t)D_FF * D_MODEL * 2;
    unsigned short* ff1b  = (unsigned short*)ws; ws += (size_t)M_ROWS * D_FF * 2;
    unsigned short* wff2T = (unsigned short*)ws; ws += (size_t)D_MODEL * D_FF * 2;

    auto tgrid = [](size_t elems) { return dim3((unsigned)((elems + 255) / 256)); };
    transpose_to_bf16_kernel<<<tgrid((size_t)D_MODEL*3*D_MODEL), 256, 0, stream>>>(
        w_qkv, wqkvT, D_MODEL, 3*D_MODEL);
    transpose_to_bf16_kernel<<<tgrid((size_t)D_MODEL*D_MODEL), 256, 0, stream>>>(
        w_out, woutT, D_MODEL, D_MODEL);
    transpose_to_bf16_kernel<<<tgrid((size_t)D_MODEL*D_FF), 256, 0, stream>>>(
        w_ff1, wff1T, D_MODEL, D_FF);
    transpose_to_bf16_kernel<<<tgrid((size_t)D_FF*D_MODEL), 256, 0, stream>>>(
        w_ff2, wff2T, D_FF, D_MODEL);

    // h1 = LN1(x)
    layernorm_bf16_kernel<<<M_ROWS, 256, 0, stream>>>(x, ln1_g, ln1_b, h1, D_MODEL);

    // qkv = h1 @ w_qkv + b_qkv   (bf16 out)
    gemm_wmma_kernel<0><<<dim3(3*D_MODEL/128, M_ROWS/64), 128, 0, stream>>>(
        h1, wqkvT, b_qkv, nullptr, qkvb, M_ROWS, 3*D_MODEL, D_MODEL);

    // attn = causal softmax(QK^T/8) V
    attention_kernel<<<M_ROWS * N_HEADS / (16 * 4), 128, 0, stream>>>(qkvb, attnb);

    // x1 = x + attn @ w_out + b_out   (fp32 out)
    gemm_wmma_kernel<2><<<dim3(D_MODEL/128, M_ROWS/64), 128, 0, stream>>>(
        attnb, woutT, b_out, x, x1, M_ROWS, D_MODEL, D_MODEL);

    // h2 = LN2(x1)
    layernorm_bf16_kernel<<<M_ROWS, 256, 0, stream>>>(x1, ln2_g, ln2_b, h2, D_MODEL);

    // ff1 = gelu(h2 @ w_ff1 + b_ff1)   (bf16 out)
    gemm_wmma_kernel<1><<<dim3(D_FF/128, M_ROWS/64), 128, 0, stream>>>(
        h2, wff1T, b_ff1, nullptr, ff1b, M_ROWS, D_FF, D_MODEL);

    // out = x1 + ff1 @ w_ff2 + b_ff2   (fp32 out)
    gemm_wmma_kernel<2><<<dim3(D_MODEL/128, M_ROWS/64), 128, 0, stream>>>(
        ff1b, wff2T, b_ff2, x1, (float*)d_out, M_ROWS, D_MODEL, D_FF);
}
